// STHMLayer_44684839748142
// MI455X (gfx1250) — matrix operand order repaired and checked by reading, not compiled, not run on hardware
//
#include <hip/hip_runtime.h>
#include <math.h>

#define D_MODEL 256
#define NWAVES  8

typedef __attribute__((ext_vector_type(16))) __bf16 v16bf;
typedef __attribute__((ext_vector_type(8)))  float  v8f;

union FragB16 { v16bf v; __bf16 h[16]; uint4 q[2]; };
union Pack8   { uint4 q; __bf16 h[8]; };

__device__ __forceinline__ v8f v8f_zero() {
  v8f z = {0.f, 0.f, 0.f, 0.f, 0.f, 0.f, 0.f, 0.f};
  return z;
}

__device__ __forceinline__ float gelu_exact(float x) {
  return 0.5f * x * (1.0f + erff(x * 0.70710678118654752440f));
}

// CDNA5 async global->LDS copy (16B), tracked by ASYNCcnt (CDNA5 ISA).
__device__ __forceinline__ void async_copy_b128(unsigned lds_off, const void* gaddr) {
  asm volatile("global_load_async_to_lds_b128 %0, %1, off"
               :: "v"(lds_off), "v"(gaddr) : "memory");
}
__device__ __forceinline__ void async_wait0() {
  asm volatile("s_wait_asynccnt 0x0" ::: "memory");
}

// ---------------------------------------------------------------------------
// fp32 -> bf16 bulk convert (8 elems/thread, 16B stores)
// ---------------------------------------------------------------------------
__global__ __launch_bounds__(256)
void cvt_f32_bf16_kernel(const float* __restrict__ in, __bf16* __restrict__ out,
                         long long n8)
{
  long long t = (long long)blockIdx.x * 256 + threadIdx.x;
  if (t >= n8) return;
  float4 a = *(const float4*)(in + t * 8);
  float4 b = *(const float4*)(in + t * 8 + 4);
  Pack8 p;
  p.h[0] = (__bf16)a.x; p.h[1] = (__bf16)a.y; p.h[2] = (__bf16)a.z; p.h[3] = (__bf16)a.w;
  p.h[4] = (__bf16)b.x; p.h[5] = (__bf16)b.y; p.h[6] = (__bf16)b.z; p.h[7] = (__bf16)b.w;
  *(uint4*)(out + t * 8) = p.q;
}

// ---------------------------------------------------------------------------
// Weight pre-pass: W[K][N] fp32 -> Wt[N][K] bf16 (done once; weights sit in L2)
// ---------------------------------------------------------------------------
__global__ __launch_bounds__(256)
void wt_transpose_kernel(const float* __restrict__ W, __bf16* __restrict__ Wt,
                         int K, int N)
{
  int idx = blockIdx.x * 256 + threadIdx.x;
  if (idx >= K * N) return;
  int k = idx / N, n = idx - k * N;           // coalesced read along n
  Wt[(size_t)n * K + k] = (__bf16)W[idx];
}

// ---------------------------------------------------------------------------
// GEMM: out[M,N] = act(A[M,K]bf16 @ Wt[N,K]bf16^T + bias) (+res)
// Double-buffered LDS staging via GLOBAL_LOAD_ASYNC_TO_LDS_B128: tile k+1 DMA
// overlaps WMMA on tile k; one s_wait_asynccnt + one barrier per k-step.
// Block 128x128x32, 8 waves (4Mx2N), wave tile 32x64 -> 8 WMMA accumulators.
// OMODE: 0=f32 out, 1=bf16 out.  RMODE: 0=none, 1=f32 res, 2=bf16 res.
// ---------------------------------------------------------------------------
#define BM 128
#define BN 128
#define BK 32
#define LDT 40   // padded K-stride in halfwords (80B rows, 16B aligned)

template<int OMODE, int RMODE>
__global__ __launch_bounds__(256)
void gemm_bf16_kernel(const __bf16* __restrict__ A, const __bf16* __restrict__ Wt,
                      const float* __restrict__ bias, const void* __restrict__ res,
                      void* __restrict__ out, int M, int N, int K, int act)
{
  __shared__ __align__(16) __bf16 ldsA[2][BM][LDT];   // [buf][m][k]
  __shared__ __align__(16) __bf16 ldsB[2][BN][LDT];   // [buf][n][k]

  const int tid  = threadIdx.x;
  const int lane = tid & 31;
  const int wid  = tid >> 5;
  const int half = lane >> 4;
  const int l16  = lane & 15;

  const int n0 = blockIdx.x * BN;
  const int m0 = blockIdx.y * BM;

  const int waveM = wid & 3;
  const int waveN = wid >> 2;

  // stage one 128x32 A tile + 128x32 B tile into buffer `buf` (4 async/thread)
  auto stage = [&](int k0, int buf) {
#pragma unroll
    for (int i = 0; i < 2; ++i) {
      int linear = tid + 256 * i;          // 0..511
      int row = linear >> 2;               // 4 chunks per row
      int c   = (linear & 3) << 3;         // halfword col: 0,8,16,24
      int gr  = m0 + row;
      if (gr >= M) gr = M - 1;             // clamp: keeps async issue EXEC-uniform
      async_copy_b128((unsigned)(size_t)&ldsA[buf][row][c], A + (size_t)gr * K + k0 + c);
      async_copy_b128((unsigned)(size_t)&ldsB[buf][row][c], Wt + (size_t)(n0 + row) * K + k0 + c);
    }
  };

  v8f acc[2][4];
#pragma unroll
  for (int mt = 0; mt < 2; ++mt)
#pragma unroll
    for (int nt = 0; nt < 4; ++nt)
      acc[mt][nt] = v8f_zero();

  // prologue: tile 0 into buffer 0
  stage(0, 0);
  async_wait0();
  __syncthreads();

  const int KT = K / BK;
  for (int kt = 0; kt < KT; ++kt) {
    const int p = kt & 1;
    if (kt + 1 < KT) stage((kt + 1) * BK, p ^ 1);   // DMA next tile, overlapped

    FragB16 fa[2], fb[4];
#pragma unroll
    for (int mt = 0; mt < 2; ++mt) {
      int m = waveM * 32 + mt * 16 + l16;
      fa[mt].q[0] = *(const uint4*)&ldsA[p][m][8 * half];        // K = 8h..8h+7
      fa[mt].q[1] = *(const uint4*)&ldsA[p][m][16 + 8 * half];   // K = 16+8h..+7
    }
#pragma unroll
    for (int nt = 0; nt < 4; ++nt) {
      int n = waveN * 64 + nt * 16 + l16;
      fb[nt].q[0] = *(const uint4*)&ldsB[p][n][8 * half];        // K = 8h..8h+7
      fb[nt].q[1] = *(const uint4*)&ldsB[p][n][16 + 8 * half];   // K = 16+8h..+7
    }
#pragma unroll
    for (int mt = 0; mt < 2; ++mt)
#pragma unroll
      for (int nt = 0; nt < 4; ++nt)
        acc[mt][nt] = __builtin_amdgcn_wmma_f32_16x16x32_bf16(
            false, fa[mt].v, false, fb[nt].v, (short)0, acc[mt][nt], false, false);

    if (kt + 1 < KT) async_wait0();   // next tile landed (in-order completion)
    __syncthreads();                  // all waves done reading buf p; next tile visible
  }

  // epilogue: bias (+gelu) (+residual), fp32 math
#pragma unroll
  for (int mt = 0; mt < 2; ++mt) {
#pragma unroll
    for (int nt = 0; nt < 4; ++nt) {
      int col = n0 + waveN * 64 + nt * 16 + l16;
      float bv = bias[col];
#pragma unroll
      for (int r = 0; r < 8; ++r) {
        int row = m0 + waveM * 32 + mt * 16 + r + 8 * half;
        if (row < M) {
          float val = acc[mt][nt][r] + bv;
          if (act == 1) val = gelu_exact(val);
          if (RMODE == 1) val += ((const float*)res)[(size_t)row * N + col];
          if (RMODE == 2) val += (float)((const __bf16*)res)[(size_t)row * N + col];
          if (OMODE == 0) ((float*)out)[(size_t)row * N + col] = val;
          else            ((__bf16*)out)[(size_t)row * N + col] = (__bf16)val;
        }
      }
    }
  }
}

// ---------------------------------------------------------------------------
// Fused Hopfield attention per (batch, head), bf16 in/out, E=32.
// scores = Q K^T * scale (WMMA, fragments straight from global bf16),
// sparsemax in accumulator registers (Michelot), out = P @ V (WMMA).
// ---------------------------------------------------------------------------
template<int SPAD, int PKC>
__global__ __launch_bounds__(256)
void hopfield_attn_kernel(const __bf16* __restrict__ Q, const __bf16* __restrict__ Kp,
                          const __bf16* __restrict__ V, __bf16* __restrict__ Out,
                          int L, int S, long long kv_batch_stride, float scale)
{
  constexpr int NT  = SPAD / 16;
  constexpr int NKC = PKC / 32;
  constexpr int PST = PKC + 8;

  __shared__ __align__(16) __bf16 sP[128][PST];   // probabilities
  __shared__ __align__(16) __bf16 sVt[32][PST];   // V transposed [e][s]

  const int tid  = threadIdx.x;
  const int lane = tid & 31;
  const int wid  = tid >> 5;
  const int half = lane >> 4;
  const int l16  = lane & 15;

  const int head  = blockIdx.x;
  const int batch = blockIdx.y;

  const size_t qbase  = ((size_t)batch * L) * D_MODEL + head * 32;
  const size_t kvbase = (size_t)batch * kv_batch_stride + head * 32;

  // phase 1: V^T into LDS, zero beyond S
  for (int idx = tid; idx < 32 * PKC; idx += 256) {
    int e  = idx & 31;
    int kp = idx >> 5;
    sVt[e][kp] = (kp < S) ? V[kvbase + (size_t)kp * D_MODEL + e] : (__bf16)0.0f;
  }

  const int nstrips = L >> 4;

  // phase 2: scores + in-register sparsemax + write P (wave-private strips)
  for (int strip = wid; strip < nstrips; strip += NWAVES) {
    FragB16 fa;   // Q fragment: two 8-elem K groups, direct bf16 vector loads
    {
      int m = strip * 16 + l16;
      const __bf16* qrow = Q + qbase + (size_t)m * D_MODEL;
      fa.q[0] = *(const uint4*)(qrow + 8 * half);
      fa.q[1] = *(const uint4*)(qrow + 16 + 8 * half);
    }
    v8f acc[NT];
#pragma unroll
    for (int nt = 0; nt < NT; ++nt) {
      FragB16 fb;   // B = K^T: lane reads e = 8h..8h+7 and 16+8h..+7 of K-row s
      int s = nt * 16 + l16;
      if (s < S) {
        const __bf16* krow = Kp + kvbase + (size_t)s * D_MODEL + 8 * half;
        fb.q[0] = *(const uint4*)(krow);
        fb.q[1] = *(const uint4*)(krow + 16);
      } else {
        fb.q[0] = make_uint4(0u, 0u, 0u, 0u);
        fb.q[1] = make_uint4(0u, 0u, 0u, 0u);
      }
      acc[nt] = __builtin_amdgcn_wmma_f32_16x16x32_bf16(
          false, fa.v, false, fb.v, (short)0, v8f_zero(), false, false);
    }

    // sparsemax: VGPR slot r holds row (strip*16 + r + 8*half); reduce per half
#pragma unroll 1
    for (int r = 0; r < 8; ++r) {
      float z[NT];
      bool  ba[NT];
      float sum = 0.f; int cnt = 0;
#pragma unroll
      for (int nt = 0; nt < NT; ++nt) {
        z[nt] = acc[nt][r] * scale;
        bool a = (nt * 16 + l16) < S;
        ba[nt] = a;
        if (a) { sum += z[nt]; cnt++; }
      }
#pragma unroll
      for (int off = 1; off < 16; off <<= 1) {
        sum += __shfl_xor(sum, off, 32);
        cnt += __shfl_xor(cnt, off, 32);
      }
      float tau = (sum - 1.0f) / (float)cnt;
      for (int it = 0; it < 32; ++it) {      // Michelot: prune z <= tau
        float ns = 0.f; int nc = 0;
#pragma unroll
        for (int nt = 0; nt < NT; ++nt) {
          ba[nt] = ba[nt] && (z[nt] > tau);
          if (ba[nt]) { ns += z[nt]; nc++; }
        }
#pragma unroll
        for (int off = 1; off < 16; off <<= 1) {
          ns += __shfl_xor(ns, off, 32);
          nc += __shfl_xor(nc, off, 32);
        }
        if (nc == cnt) break;
        cnt = nc;
        tau = (ns - 1.0f) / (float)nc;
      }
      int m = strip * 16 + r + 8 * half;
#pragma unroll
      for (int nt = 0; nt < NT; ++nt) {
        float p = ba[nt] ? (z[nt] - tau) : 0.0f;
        sP[m][nt * 16 + l16] = (__bf16)p;
      }
    }
    if (SPAD < PKC) {   // zero K-chunk pad columns of this strip's rows
      int m = strip * 16 + l16;
      for (int c = SPAD + half; c < PKC; c += 2)
        sP[m][c] = (__bf16)0.0f;
    }
  }

  __syncthreads();   // publish sVt for the out-GEMM (sP rows are wave-private)

  // phase 3: out = P @ V
  for (int strip = wid; strip < nstrips; strip += NWAVES) {
    v8f oacc[2];
    oacc[0] = v8f_zero(); oacc[1] = v8f_zero();
#pragma unroll
    for (int kc = 0; kc < NKC; ++kc) {
      FragB16 fpa;
      int m = strip * 16 + l16;
      fpa.q[0] = *(const uint4*)&sP[m][kc * 32 + 8 * half];
      fpa.q[1] = *(const uint4*)&sP[m][kc * 32 + 16 + 8 * half];
#pragma unroll
      for (int nt = 0; nt < 2; ++nt) {
        FragB16 fvb;
        int e = nt * 16 + l16;
        fvb.q[0] = *(const uint4*)&sVt[e][kc * 32 + 8 * half];
        fvb.q[1] = *(const uint4*)&sVt[e][kc * 32 + 16 + 8 * half];
        oacc[nt] = __builtin_amdgcn_wmma_f32_16x16x32_bf16(
            false, fpa.v, false, fvb.v, (short)0, oacc[nt], false, false);
      }
    }
#pragma unroll
    for (int nt = 0; nt < 2; ++nt) {
      int e = nt * 16 + l16;
#pragma unroll
      for (int r = 0; r < 8; ++r) {
        int m = strip * 16 + r + 8 * half;
        Out[qbase + (size_t)m * D_MODEL + e] = (__bf16)oacc[nt][r];
      }
    }
  }
}

// ---------------------------------------------------------------------------
// LayerNorm over last dim (256), bf16 in/out, fp32 math, one wave per row.
// ---------------------------------------------------------------------------
__global__ __launch_bounds__(256)
void layernorm_kernel(const __bf16* __restrict__ in, const float* __restrict__ g,
                      const float* __restrict__ b, __bf16* __restrict__ out, int nrows)
{
  const int lane = threadIdx.x & 31;
  const int wid  = threadIdx.x >> 5;
  const int row  = blockIdx.x * NWAVES + wid;
  if (row >= nrows) return;
  const int c0 = lane * 8;
  Pack8 p;
  p.q = *(const uint4*)(in + (size_t)row * D_MODEL + c0);
  float xv[8];
  float s = 0.f, ss = 0.f;
#pragma unroll
  for (int j = 0; j < 8; ++j) {
    xv[j] = (float)p.h[j];
    s += xv[j];
    ss += xv[j] * xv[j];
  }
#pragma unroll
  for (int off = 1; off < 32; off <<= 1) {
    s  += __shfl_xor(s, off, 32);
    ss += __shfl_xor(ss, off, 32);
  }
  float mean = s * (1.0f / 256.0f);
  float var  = ss * (1.0f / 256.0f) - mean * mean;
  float inv  = rsqrtf(var + 1e-5f);
  float4 g0 = *(const float4*)(g + c0), g1 = *(const float4*)(g + c0 + 4);
  float4 b0 = *(const float4*)(b + c0), b1 = *(const float4*)(b + c0 + 4);
  const float gg[8] = {g0.x, g0.y, g0.z, g0.w, g1.x, g1.y, g1.z, g1.w};
  const float bb[8] = {b0.x, b0.y, b0.z, b0.w, b1.x, b1.y, b1.z, b1.w};
  Pack8 r;
#pragma unroll
  for (int j = 0; j < 8; ++j)
    r.h[j] = (__bf16)((xv[j] - mean) * inv * gg[j] + bb[j]);
  *(uint4*)(out + (size_t)row * D_MODEL + c0) = r.q;
}

// ---------------------------------------------------------------------------
// (B, d1, d2, 256) -> (B, d2, d1, 256) row-vector transposes
// ---------------------------------------------------------------------------
__global__ __launch_bounds__(64)
void transpose_mid_bf16(const __bf16* __restrict__ in, __bf16* __restrict__ out,
                        int d1, int d2)
{
  const int row = blockIdx.x;
  const int j   = row % d2;
  const int t   = row / d2;
  const int i   = t % d1;
  const int bb  = t / d1;
  const uint2* src = (const uint2*)(in + (size_t)row * D_MODEL);
  uint2* dst = (uint2*)(out + (((size_t)bb * d2 + j) * d1 + i) * D_MODEL);
  dst[threadIdx.x] = src[threadIdx.x];
}

__global__ __launch_bounds__(64)
void transpose_mid_f32(const float* __restrict__ in, float* __restrict__ out,
                       int d1, int d2)
{
  const int row = blockIdx.x;
  const int j   = row % d2;
  const int t   = row / d2;
  const int i   = t % d1;
  const int bb  = t / d1;
  const float4* src = (const float4*)(in + (size_t)row * D_MODEL);
  float4* dst = (float4*)(out + (((size_t)bb * d2 + j) * d1 + i) * D_MODEL);
  dst[threadIdx.x] = src[threadIdx.x];
}

// ---------------------------------------------------------------------------
// Host orchestration
// ---------------------------------------------------------------------------
static void launch_gemm(const __bf16* A, const __bf16* Wt, const float* bias,
                        const void* res, void* out, int M, int N, int K,
                        int act, int omode, int rmode, hipStream_t stream) {
  dim3 grid(N / BN, (M + BM - 1) / BM);
  if (omode == 1 && rmode == 0)
    gemm_bf16_kernel<1, 0><<<grid, 256, 0, stream>>>(A, Wt, bias, res, out, M, N, K, act);
  else if (omode == 1 && rmode == 1)
    gemm_bf16_kernel<1, 1><<<grid, 256, 0, stream>>>(A, Wt, bias, res, out, M, N, K, act);
  else if (omode == 1 && rmode == 2)
    gemm_bf16_kernel<1, 2><<<grid, 256, 0, stream>>>(A, Wt, bias, res, out, M, N, K, act);
  else
    gemm_bf16_kernel<0, 2><<<grid, 256, 0, stream>>>(A, Wt, bias, res, out, M, N, K, act);
}

static void launch_attn(const __bf16* Q, const __bf16* K, const __bf16* V, __bf16* O,
                        int nb, int L, int S, long long kvstride, hipStream_t stream) {
  const float scale = 1.0f / sqrtf(32.0f);
  dim3 grid(8, nb);   // (head, batch)
  if (S == 128)
    hopfield_attn_kernel<128, 128><<<grid, 256, 0, stream>>>(Q, K, V, O, L, S, kvstride, scale);
  else if (S == 64)
    hopfield_attn_kernel<64, 64><<<grid, 256, 0, stream>>>(Q, K, V, O, L, S, kvstride, scale);
  else
    hopfield_attn_kernel<16, 32><<<grid, 256, 0, stream>>>(Q, K, V, O, L, S, kvstride, scale);
}

extern "C" void kernel_launch(void* const* d_in, const int* in_sizes, int n_in,
                              void* d_out, int out_size, void* d_ws, size_t ws_size,
                              hipStream_t stream)
{
  (void)in_sizes; (void)n_in; (void)out_size; (void)ws_size;

  const float* x     = (const float*)d_in[0];
  const float* ct_wq = (const float*)d_in[1];  const float* ct_bq = (const float*)d_in[2];
  const float* ct_wk = (const float*)d_in[3];  const float* ct_bk = (const float*)d_in[4];
  const float* ct_wv = (const float*)d_in[5];  const float* ct_bv = (const float*)d_in[6];
  const float* ct_wo = (const float*)d_in[7];  const float* ct_bo = (const float*)d_in[8];
  const float* cs_wq = (const float*)d_in[9];  const float* cs_bq = (const float*)d_in[10];
  const float* cs_wk = (const float*)d_in[11]; const float* cs_bk = (const float*)d_in[12];
  const float* cs_wv = (const float*)d_in[13]; const float* cs_bv = (const float*)d_in[14];
  const float* cs_wo = (const float*)d_in[15]; const float* cs_bo = (const float*)d_in[16];
  const float* hp_wq = (const float*)d_in[17]; const float* hp_bq = (const float*)d_in[18];
  const float* hp_wk = (const float*)d_in[19]; const float* hp_bk = (const float*)d_in[20];
  const float* hp_wv = (const float*)d_in[21]; const float* hp_bv = (const float*)d_in[22];
  const float* hp_wo = (const float*)d_in[23]; const float* hp_bo = (const float*)d_in[24];
  const float* cs_key = (const float*)d_in[25];
  const float* ln_g  = (const float*)d_in[26]; const float* ln_b  = (const float*)d_in[27];
  const float* m1_w1 = (const float*)d_in[28]; const float* m1_b1 = (const float*)d_in[29];
  const float* m1_w2 = (const float*)d_in[30]; const float* m1_b2 = (const float*)d_in[31];
  const float* m2_w1 = (const float*)d_in[32]; const float* m2_b1 = (const float*)d_in[33];
  const float* m2_w2 = (const float*)d_in[34]; const float* m2_b2 = (const float*)d_in[35];

  float* out = (float*)d_out;
  char*  wsb = (char*)d_ws;

  const size_t R = (size_t)32768 * 256;
  size_t o = 0;
  __bf16* xb = (__bf16*)(wsb + o); o += R * 2;
  __bf16* b0 = (__bf16*)(wsb + o); o += R * 2;
  __bf16* b1 = (__bf16*)(wsb + o); o += R * 2;
  __bf16* b2 = (__bf16*)(wsb + o); o += R * 2;
  __bf16* b3 = (__bf16*)(wsb + o); o += R * 2;
  __bf16* b4 = (__bf16*)(wsb + o); o += R * 2;
  __bf16* b5 = (__bf16*)(wsb + o); o += (size_t)32768 * 1024 * 2;   // DFF buffer
  float*  fin = (float*)(wsb + o); o += R * 4;                      // final pre-transpose

  // transposed bf16 weight copies
  const float* wsrc[16] = { ct_wq, ct_wk, ct_wv, ct_wo,
                            cs_wq, cs_wk, cs_wv, cs_wo,
                            hp_wq, hp_wk, hp_wv, hp_wo,
                            m1_w1, m1_w2, m2_w1, m2_w2 };
  const int wkk[16] = {256,256,256,256, 256,256,256,256, 256,256,256,256, 256,1024,256,1024};
  const int wnn[16] = {256,256,256,256, 256,256,256,256, 256,256,256,256, 1024,256,1024,256};
  __bf16* wt[16];
  for (int i = 0; i < 16; ++i) {
    wt[i] = (__bf16*)(wsb + o);
    o += (size_t)wkk[i] * wnn[i] * 2;
  }
  __bf16* keyb = (__bf16*)(wsb + o); o += 2560 * 2;
  __bf16* skp  = (__bf16*)(wsb + o); o += 16 * 256 * 2;
  __bf16* sv   = (__bf16*)(wsb + o); o += 16 * 256 * 2;

  const int M = 32768, D = 256, F = 1024;

  // ---- one-time precision/layout pre-passes ----
  cvt_f32_bf16_kernel<<<dim3((unsigned)((R / 8 + 255) / 256)), 256, 0, stream>>>(x, xb, (long long)(R / 8));
  cvt_f32_bf16_kernel<<<dim3(2), 256, 0, stream>>>(cs_key, keyb, 2560 / 8);
  for (int i = 0; i < 16; ++i) {
    int tot = wkk[i] * wnn[i];
    wt_transpose_kernel<<<dim3((tot + 255) / 256), 256, 0, stream>>>(wsrc[i], wt[i], wkk[i], wnn[i]);
  }

  // ---- stage 1: time hopfield (256 batches, L=S=128) ----
  launch_gemm(xb, wt[0], ct_bq, nullptr, b0, M, D, D, 0, 1, 0, stream);     // Q
  launch_gemm(xb, wt[1], ct_bk, nullptr, b1, M, D, D, 0, 1, 0, stream);     // Kproj
  launch_gemm(b1, wt[2], ct_bv, nullptr, b2, M, D, D, 0, 1, 0, stream);     // V(Kproj)
  launch_attn(b0, b1, b2, b3, 256, 128, 128, 128 * 256, stream);
  launch_gemm(b3, wt[3], ct_bo, x, b4, M, D, D, 0, 1, 1, stream);           // dim_in = x + attn
  launch_gemm(b4, wt[12], m1_b1, nullptr, b5, M, F, D, 1, 1, 0, stream);    // gelu MLP1
  launch_gemm(b5, wt[13], m1_b2, b4, b0, M, D, F, 0, 1, 2, stream);         // + residual

  // ---- (b ts seg d) -> (b seg ts d) ----
  transpose_mid_bf16<<<dim3(32768), 64, 0, stream>>>(b0, b1, 64, 128);

  // ---- stage 2: cs hopfield (512 batches, L=64, S=10, shared K/V) ----
  launch_gemm(b1, wt[4], cs_bq, nullptr, b2, M, D, D, 0, 1, 0, stream);     // Q
  launch_gemm(keyb, wt[5], cs_bk, nullptr, skp, 10, D, D, 0, 1, 0, stream); // Kproj
  launch_gemm(skp, wt[6], cs_bv, nullptr, sv, 10, D, D, 0, 1, 0, stream);   // V(Kproj)
  launch_attn(b2, skp, sv, b3, 512, 64, 10, 0, stream);
  launch_gemm(b3, wt[7], cs_bo, nullptr, b4, M, D, D, 0, 1, 0, stream);     // series_h

  // ---- stage 3: hp hopfield (512 batches, L=S=64) ----
  launch_gemm(b1, wt[8], hp_bq, nullptr, b0, M, D, D, 0, 1, 0, stream);     // Q
  launch_gemm(b1, wt[9], hp_bk, nullptr, b2, M, D, D, 0, 1, 0, stream);     // Kproj
  launch_gemm(b2, wt[10], hp_bv, nullptr, b3, M, D, D, 0, 1, 0, stream);    // V(Kproj)
  launch_attn(b0, b2, b3, b5, 512, 64, 64, 64 * 256, stream);
  launch_gemm(b5, wt[11], hp_bo, b4, b1, M, D, D, 0, 1, 2, stream);         // series_h + pooled_h

  // ---- LayerNorm + MLP2 ----
  layernorm_kernel<<<dim3(32768 / NWAVES), 256, 0, stream>>>(b1, ln_g, ln_b, b0, 32768);
  launch_gemm(b0, wt[14], m2_b1, nullptr, b5, M, F, D, 1, 1, 0, stream);    // gelu MLP2
  launch_gemm(b5, wt[15], m2_b2, b0, fin, M, D, F, 0, 0, 2, stream);        // + residual, f32 out

  // ---- (b seg ts d) -> (b ts seg d) ----
  transpose_mid_f32<<<dim3(32768), 64, 0, stream>>>(fin, out, 128, 64);
}
